// OptimalTransportAlign_1932735283710
// MI455X (gfx1250) — matrix operand order repaired and checked by reading, not compiled, not run on hardware
//
#include <hip/hip_runtime.h>
#include <hip/hip_bf16.h>
#include <stdint.h>

// ---------------- types ----------------
typedef float  v2f   __attribute__((ext_vector_type(2)));
typedef float  v8f   __attribute__((ext_vector_type(8)));
typedef __bf16 v16bf __attribute__((ext_vector_type(16)));
typedef unsigned int v4u __attribute__((ext_vector_type(4)));
typedef int    v8i   __attribute__((ext_vector_type(8)));
typedef int    v4i   __attribute__((ext_vector_type(4)));

// ---------------- problem dims ----------------
#define MW    1024          // B*SW rows of x
#define ML    512           // B*SL rows of y
#define DPROJ 256
#define KW    51865
#define KL    32000
#define VOUT  51865
#define EPS      0.0025f
#define INV_EPS  400.0f
#define N_ITERS  50

// ---------------- GEMM tiling ----------------
#define BLK_M 128
#define BLK_N 256
#define KSTEP 32
#define GTHREADS 512        // 16 waves: 2 (M) x 8 (N)
#define ASTR 36             // padded LDS row stride for A (floats): TDM pads 32->36
#define BSTR 258            // padded LDS row stride for B (floats): TDM pads 256->258
#define A_ELEMS (BLK_M*ASTR)
#define B_ELEMS (KSTEP*BSTR)
#define GEMM_LDS_BYTES ((2*A_ELEMS + 2*B_ELEMS)*4)

#if defined(__has_include)
#  if __has_include(<hip/amd_detail/amd_gfx1250_TDM.h>)
#    define TDM_6ARG 1
#  endif
#endif

// -------------------------------------------------------------------------
// TDM helpers
// -------------------------------------------------------------------------
__device__ __forceinline__ uint32_t lds_off_of(const void* p) {
  // low 32 bits of a flat shared-aperture address == LDS byte offset
  return (uint32_t)(uintptr_t)p;
}

// 2D tile load: tile_w x tile_h elements (f32), row stride `stride_elems`,
// OOB (beyond rem_w/rem_h from tile start) zero-filled by the TDM.
__device__ __forceinline__ void tdm_load_2d(uint32_t lds_off, const float* gp,
                                            uint32_t tile_w, uint32_t tile_h,
                                            long rem_w, long rem_h,
                                            uint32_t stride_elems, uint32_t pad_bits)
{
  uint64_t ga  = (uint64_t)(uintptr_t)gp;
  uint32_t td0 = rem_w <= 0 ? 0u : (uint32_t)rem_w;
  uint32_t td1 = rem_h <= 0 ? 0u : (uint32_t)rem_h;
  v4u g0;
  g0.x = 1u;                                             // count=1 (valid user D#)
  g0.y = lds_off;                                        // lds_addr
  g0.z = (uint32_t)(ga & 0xffffffffu);                   // global_addr[31:0]
  g0.w = (uint32_t)((ga >> 32) & 0x01ffffffu) | (2u<<30);// global_addr[56:32] | type=2
  v8i g1;
  g1[0] = (int)((2u << 16) | pad_bits);                  // data_size=4B | pad cfg
  g1[1] = (int)((td0 & 0xffffu) << 16);                  // tensor_dim0[15:0]
  g1[2] = (int)((td0 >> 16) | ((td1 & 0xffffu) << 16));  // dim0 hi | dim1 lo
  g1[3] = (int)((td1 >> 16) | (tile_w << 16));           // dim1 hi | tile_dim0
  g1[4] = (int)(tile_h & 0xffffu);                       // tile_dim1 (tile_dim2=0)
  g1[5] = (int)stride_elems;                             // tensor_dim0_stride[31:0]
  g1[6] = 0;
  g1[7] = 0;
  v4i z4 = {0,0,0,0};
#ifdef TDM_6ARG
  v8i z8 = {0,0,0,0,0,0,0,0};
  __builtin_amdgcn_tensor_load_to_lds(g0, g1, z4, z4, z8, 0);
#else
  __builtin_amdgcn_tensor_load_to_lds(g0, g1, z4, z4, 0);
#endif
}

// pad config: A rows 32 dwords -> +4 dwords (interval code 4 = 32 dw, amount code 3 = 4 dw)
#define APAD ((1u<<20) | (4u<<22) | (3u<<25))
// pad config: B rows 256 dwords -> +2 dwords (interval code 7 = 256 dw, amount code 1 = 2 dw)
#define BPAD ((1u<<20) | (7u<<22) | (1u<<25))

// -------------------------------------------------------------------------
// Generic bf16-WMMA GEMM: Out(+)= X[M,K] @ W[K,N] (+bias). Split-K via grid.z.
// -------------------------------------------------------------------------
__global__ void __launch_bounds__(GTHREADS)
gemm_bf16_kernel(const float* __restrict__ X, const float* __restrict__ W,
                 const float* __restrict__ bias, float* __restrict__ Out,
                 int M, int N, int K, int kChunk, int useAtomic)
{
  extern __shared__ float smem[];
  // layout: [Abuf0 | Abuf1 | Bbuf0 | Bbuf1] -- no pointer arrays (LDS addrspacecast
  // in a static aggregate initializer is not linkable); use integer offsets instead.

  const int mBase  = blockIdx.y * BLK_M;
  const int nBase  = blockIdx.x * BLK_N;
  const int kBegin = blockIdx.z * kChunk;
  int kEnd = kBegin + kChunk; if (kEnd > K) kEnd = K;
  const int nsteps = (kEnd > kBegin) ? ((kEnd - kBegin + KSTEP - 1) / KSTEP) : 0;
  if (nsteps <= 0) return;

  const int lane = threadIdx.x & 31;
  const int wave = threadIdx.x >> 5;
  const int wm = wave >> 3;   // 0..1 : 64-row group
  const int wn = wave & 7;    // 0..7 : 32-col group
  const int lh = lane >> 4;   // lane half
  const int ll = lane & 15;

  const uint32_t smemOff = lds_off_of(smem);

  if (threadIdx.x < 32) {     // wave 0 drives the TDM
    tdm_load_2d(smemOff, X + (size_t)mBase*K + kBegin, KSTEP, BLK_M,
                (long)K - kBegin, (long)M - mBase, (uint32_t)K, APAD);
    tdm_load_2d(smemOff + 2u*A_ELEMS*4u, W + (size_t)kBegin*N + nBase, BLK_N, KSTEP,
                (long)N - nBase, (long)K - kBegin, (uint32_t)N, BPAD);
  }

  v8f acc[4][2];
  #pragma unroll
  for (int i = 0; i < 4; ++i)
    #pragma unroll
    for (int j = 0; j < 2; ++j)
      acc[i][j] = (v8f){0.f,0.f,0.f,0.f,0.f,0.f,0.f,0.f};

  for (int s = 0; s < nsteps; ++s) {
    if ((s + 1) < nsteps) {
      if (threadIdx.x < 32) {
        int k0 = kBegin + (s + 1) * KSTEP;
        uint32_t b = (uint32_t)((s + 1) & 1);
        tdm_load_2d(smemOff + b*(uint32_t)A_ELEMS*4u,
                    X + (size_t)mBase*K + k0, KSTEP, BLK_M,
                    (long)K - k0, (long)M - mBase, (uint32_t)K, APAD);
        tdm_load_2d(smemOff + 2u*A_ELEMS*4u + b*(uint32_t)B_ELEMS*4u,
                    W + (size_t)k0*N + nBase, BLK_N, KSTEP,
                    (long)N - nBase, (long)K - k0, (uint32_t)N, BPAD);
      }
      __builtin_amdgcn_s_wait_tensorcnt(2);   // current buffer complete (in-order)
    } else {
      __builtin_amdgcn_s_wait_tensorcnt(0);
    }
    __syncthreads();

    const int buf = s & 1;
    const float* Ab = smem + buf * A_ELEMS;
    const float* Bb = smem + 2 * A_ELEMS + buf * B_ELEMS;

    // A fragments (16x32 bf16): lanes 0-15 -> K 0..7,16..23 ; lanes 16-31 -> K 8..15,24..31
    v16bf af[4];
    #pragma unroll
    for (int mt = 0; mt < 4; ++mt) {
      const float* p = Ab + (wm*64 + mt*16 + ll) * ASTR + lh*8;
      float4 c0 = *(const float4*)(p);
      float4 c1 = *(const float4*)(p + 4);
      float4 c2 = *(const float4*)(p + 16);
      float4 c3 = *(const float4*)(p + 20);
      v16bf a;
      a[0]=(__bf16)c0.x;  a[1]=(__bf16)c0.y;  a[2]=(__bf16)c0.z;  a[3]=(__bf16)c0.w;
      a[4]=(__bf16)c1.x;  a[5]=(__bf16)c1.y;  a[6]=(__bf16)c1.z;  a[7]=(__bf16)c1.w;
      a[8]=(__bf16)c2.x;  a[9]=(__bf16)c2.y;  a[10]=(__bf16)c2.z; a[11]=(__bf16)c2.w;
      a[12]=(__bf16)c3.x; a[13]=(__bf16)c3.y; a[14]=(__bf16)c3.z; a[15]=(__bf16)c3.w;
      af[mt] = a;
    }
    // B fragments (32x16 bf16): lanes 0-15 -> K 0..7,16..23 ; lanes 16-31 -> K 8..15,24..31
    v16bf bfr[2];
    #pragma unroll
    for (int nt = 0; nt < 2; ++nt) {
      const float* p = Bb + lh*8*BSTR + (wn*32 + nt*16 + ll);
      v16bf b;
      #pragma unroll
      for (int t = 0; t < 16; ++t) b[t] = (__bf16)p[(t < 8 ? t : t + 8) * BSTR];
      bfr[nt] = b;
    }
    #pragma unroll
    for (int mt = 0; mt < 4; ++mt)
      #pragma unroll
      for (int nt = 0; nt < 2; ++nt)
        acc[mt][nt] = __builtin_amdgcn_wmma_f32_16x16x32_bf16(
            false, af[mt], false, bfr[nt], (short)0, acc[mt][nt], false, false);
    __syncthreads();
  }

  // epilogue: C/D layout -> lane: N=ll, M=r+8*lh per VGPR r
  #pragma unroll
  for (int mt = 0; mt < 4; ++mt)
    #pragma unroll
    for (int nt = 0; nt < 2; ++nt)
      #pragma unroll
      for (int r = 0; r < 8; ++r) {
        int gm = mBase + wm*64 + mt*16 + r + 8*lh;
        int gn = nBase + wn*32 + nt*16 + ll;
        if (gm < M && gn < N) {
          float v = acc[mt][nt][r];
          size_t o = (size_t)gm * N + gn;
          if (useAtomic) atomicAdd(&Out[o], v);
          else           Out[o] = v + (bias ? bias[gn] : 0.f);
        }
      }
}

// -------------------------------------------------------------------------
// init: E1 = b_whisper (bcast), E2 = b_llama, zero scalar + barrier state
// -------------------------------------------------------------------------
__global__ void init_kernel(float* E1, float* E2, const float* bw, const float* bl,
                            float* outScalar, unsigned* cnt, unsigned* gen)
{
  int gt = blockIdx.x * blockDim.x + threadIdx.x;
  int gs = gridDim.x * blockDim.x;
  for (int i = gt; i < MW*DPROJ; i += gs) E1[i] = bw[i & (DPROJ-1)];
  for (int i = gt; i < ML*DPROJ; i += gs) E2[i] = bl[i & (DPROJ-1)];
  if (gt == 0) { *outScalar = 0.f; *cnt = 0u; *gen = 0u; }
}

// -------------------------------------------------------------------------
// half squared norms: hx[i] = 0.5*||x_i||^2, hy[j] = 0.5*||y_j||^2
// -------------------------------------------------------------------------
__global__ void norms_kernel(const float* __restrict__ E1, const float* __restrict__ E2,
                             float* __restrict__ hx, float* __restrict__ hy)
{
  int r = blockIdx.x * blockDim.x + threadIdx.x;
  if (r < MW) {
    const float* p = E1 + (size_t)r * DPROJ;
    float s = 0.f;
    for (int d = 0; d < DPROJ; ++d) { float v = p[d]; s += v*v; }
    hx[r] = 0.5f * s;
  } else if (r < MW + ML) {
    int rr = r - MW;
    const float* p = E2 + (size_t)rr * DPROJ;
    float s = 0.f;
    for (int d = 0; d < DPROJ; ++d) { float v = p[d]; s += v*v; }
    hy[rr] = 0.5f * s;
  }
}

// -------------------------------------------------------------------------
// Cost matrix via f32 WMMA 16x16x4: C[r][c] = hr[r] + hc[c] - <Xr_r, Xc_c>
// One wave per 16x16 tile; x/y are L0/L2-resident so frags load from global.
// -------------------------------------------------------------------------
__global__ void __launch_bounds__(256)
cost_kernel(const float* __restrict__ Xr, const float* __restrict__ Xc,
            const float* __restrict__ hr, const float* __restrict__ hc,
            float* __restrict__ Cdst, int R, int Cc)
{
  const int lane = threadIdx.x & 31;
  const int wave = threadIdx.x >> 5;
  const int lh = lane >> 4, ll = lane & 15;
  const int tilesX = Cc >> 4;
  int tile = blockIdx.x * 8 + wave;
  int ty = tile / tilesX, tx = tile - ty * tilesX;
  if (ty >= (R >> 4)) return;

  const float* pa = Xr + (size_t)(ty*16 + ll) * DPROJ + lh*2; // A 16x4: lane half picks K pair
  const float* pb = Xc + (size_t)(tx*16 + ll) * DPROJ + lh*2; // B 4x16 mirrors A
  v8f acc = (v8f){0.f,0.f,0.f,0.f,0.f,0.f,0.f,0.f};
  #pragma unroll 8
  for (int k0 = 0; k0 < DPROJ; k0 += 4) {
    v2f a = *(const v2f*)(pa + k0);
    v2f b = *(const v2f*)(pb + k0);
    acc = __builtin_amdgcn_wmma_f32_16x16x4_f32(false, a, false, b, (short)0, acc,
                                                false, false);
  }
  float hcv = hc[tx*16 + ll];
  #pragma unroll
  for (int r = 0; r < 8; ++r) {
    int row = ty*16 + r + 8*lh;
    Cdst[(size_t)row * Cc + tx*16 + ll] = hr[row] + hcv - acc[r];
  }
}

// -------------------------------------------------------------------------
// Persistent log-domain Sinkhorn (one launch per OT problem)
// -------------------------------------------------------------------------
#define SBLOCKS 64

__device__ __forceinline__ void grid_sync(unsigned* cnt, unsigned* gen)
{
  __syncthreads();
  if (threadIdx.x == 0) {
    __threadfence();
    unsigned old = __hip_atomic_load(gen, __ATOMIC_ACQUIRE, __HIP_MEMORY_SCOPE_AGENT);
    unsigned a = __hip_atomic_fetch_add(cnt, 1u, __ATOMIC_ACQ_REL, __HIP_MEMORY_SCOPE_AGENT);
    if (a == gridDim.x - 1) {
      __hip_atomic_store(cnt, 0u, __ATOMIC_RELAXED, __HIP_MEMORY_SCOPE_AGENT);
      __hip_atomic_fetch_add(gen, 1u, __ATOMIC_ACQ_REL, __HIP_MEMORY_SCOPE_AGENT);
    } else {
      while (__hip_atomic_load(gen, __ATOMIC_ACQUIRE, __HIP_MEMORY_SCOPE_AGENT) == old)
        __builtin_amdgcn_s_sleep(2);
    }
  }
  __syncthreads();
}

// out[j] = -eps * ( LSE_i((pot[i]-Crows[j][i])/eps) - logNsrc ), one wave per j
__device__ __forceinline__ void half_update(const float* __restrict__ Crows, int nRows,
                                            int L, const float* __restrict__ pot,
                                            float logNsrc, float* __restrict__ out)
{
  int gw   = (blockIdx.x * blockDim.x + threadIdx.x) >> 5;
  int lane = threadIdx.x & 31;
  int nw   = (gridDim.x * blockDim.x) >> 5;
  for (int j = gw; j < nRows; j += nw) {
    const float* row = Crows + (size_t)j * L;
    float m = -3.0e38f, s = 0.f;
    for (int i = lane; i < L; i += 32) {
      float t = (pot[i] - row[i]) * INV_EPS;
      if (t <= m) { s += __expf(t - m); }
      else        { s = s * __expf(m - t) + 1.f; m = t; }
    }
    #pragma unroll
    for (int off = 16; off; off >>= 1) {
      float om = __shfl_xor(m, off, 32);
      float os = __shfl_xor(s, off, 32);
      float nm = fmaxf(m, om);
      s = s * __expf(m - nm) + os * __expf(om - nm);
      m = nm;
    }
    if (lane == 0) out[j] = -EPS * (m + __logf(s) - logNsrc);
  }
}

__global__ void __launch_bounds__(256)
sinkhorn_kernel(const float* __restrict__ Cf,  // [Nx x Ny] rows i
                const float* __restrict__ Cg,  // [Ny x Nx] rows j (transpose of Cf)
                float* __restrict__ f, float* __restrict__ g,
                int Nx, int Ny, float weight, float* __restrict__ outScalar,
                unsigned* cnt, unsigned* gen)
{
  int gt = blockIdx.x * blockDim.x + threadIdx.x;
  int gs = gridDim.x * blockDim.x;
  for (int i = gt; i < Nx; i += gs) f[i] = 0.f;
  for (int j = gt; j < Ny; j += gs) g[j] = 0.f;
  grid_sync(cnt, gen);

  float logNx = __logf((float)Nx), logNy = __logf((float)Ny);
  for (int it = 0; it < N_ITERS; ++it) {
    half_update(Cg, Ny, Nx, f, logNx, g);   // g from f (reduce over i)
    grid_sync(cnt, gen);
    half_update(Cf, Nx, Ny, g, logNy, f);   // f from g (reduce over j)
    grid_sync(cnt, gen);
  }

  if (blockIdx.x == 0) {
    __shared__ float red[256];
    float v = 0.f;
    for (int i = threadIdx.x; i < Nx; i += 256) v += f[i];
    v /= (float)Nx;
    float w2 = 0.f;
    for (int j = threadIdx.x; j < Ny; j += 256) w2 += g[j];
    v += w2 / (float)Ny;
    red[threadIdx.x] = v;
    __syncthreads();
    for (int o = 128; o; o >>= 1) {
      if (threadIdx.x < o) red[threadIdx.x] += red[threadIdx.x + o];
      __syncthreads();
    }
    if (threadIdx.x == 0) atomicAdd(outScalar, weight * red[0]);
  }
}

// -------------------------------------------------------------------------
// host-side launcher
// -------------------------------------------------------------------------
static inline int ksplit_chunk(int K, int targetSplits)
{
  int stepTotal   = (K + KSTEP - 1) / KSTEP;
  int stepsPerBlk = (stepTotal + targetSplits - 1) / targetSplits;
  return stepsPerBlk * KSTEP;
}

extern "C" void kernel_launch(void* const* d_in, const int* in_sizes, int n_in,
                              void* d_out, int out_size, void* d_ws, size_t ws_size,
                              hipStream_t stream)
{
  (void)in_sizes; (void)n_in; (void)out_size; (void)ws_size;
  const float* wlog = (const float*)d_in[0];
  const float* llog = (const float*)d_in[1];
  const float* Wp   = (const float*)d_in[2];
  const float* bw   = (const float*)d_in[3];
  const float* Wl   = (const float*)d_in[4];
  const float* bl   = (const float*)d_in[5];
  const float* Wo   = (const float*)d_in[6];
  const float* bo   = (const float*)d_in[7];

  float* out       = (float*)d_out;
  float* scalarOut = out + (size_t)MW * VOUT;   // transport_cost slot

  float* ws   = (float*)d_ws;
  float* E1   = ws;  ws += (size_t)MW * DPROJ;
  float* E2   = ws;  ws += (size_t)ML * DPROJ;
  float* hx   = ws;  ws += MW;
  float* hy   = ws;  ws += ML;
  float* Cxy  = ws;  ws += (size_t)MW * ML;
  float* Cyx  = ws;  ws += (size_t)ML * MW;
  float* Cxx  = ws;  ws += (size_t)MW * MW;
  float* Cyy  = ws;  ws += (size_t)ML * ML;
  float* fpot = ws;  ws += MW;
  float* gpot = ws;  ws += MW;
  unsigned* cnt = (unsigned*)ws;
  unsigned* gen = cnt + 1;

  // 0) init biases into E1/E2, zero accumulator + barrier state
  init_kernel<<<dim3(512), dim3(256), 0, stream>>>(E1, E2, bw, bl, scalarOut, cnt, gen);

  // 1) embeds (split-K bf16 WMMA, atomic accumulate onto bias)
  {
    int kc = ksplit_chunk(KW, 32);
    int zb = (KW + kc - 1) / kc;
    gemm_bf16_kernel<<<dim3(1, MW/BLK_M, zb), dim3(GTHREADS), GEMM_LDS_BYTES, stream>>>(
        wlog, Wp, nullptr, E1, MW, DPROJ, KW, kc, 1);
  }
  {
    int kc = ksplit_chunk(KL, 32);
    int zb = (KL + kc - 1) / kc;
    gemm_bf16_kernel<<<dim3(1, ML/BLK_M, zb), dim3(GTHREADS), GEMM_LDS_BYTES, stream>>>(
        llog, Wl, nullptr, E2, ML, DPROJ, KL, kc, 1);
  }

  // 2) half squared norms
  norms_kernel<<<dim3((MW + ML) / 256), dim3(256), 0, stream>>>(E1, E2, hx, hy);

  // 3) cost matrices (f32 WMMA 16x16x4); Cyx = cost(y,x) = Cxy^T; xx/yy symmetric
  cost_kernel<<<dim3((MW/16)*(ML/16)/8), dim3(256), 0, stream>>>(E1, E2, hx, hy, Cxy, MW, ML);
  cost_kernel<<<dim3((ML/16)*(MW/16)/8), dim3(256), 0, stream>>>(E2, E1, hy, hx, Cyx, ML, MW);
  cost_kernel<<<dim3((MW/16)*(MW/16)/8), dim3(256), 0, stream>>>(E1, E1, hx, hx, Cxx, MW, MW);
  cost_kernel<<<dim3((ML/16)*(ML/16)/8), dim3(256), 0, stream>>>(E2, E2, hy, hy, Cyy, ML, ML);

  // 4) debiased Sinkhorn: OT(x,y) - 0.5*OT(x,x) - 0.5*OT(y,y)
  sinkhorn_kernel<<<dim3(SBLOCKS), dim3(256), 0, stream>>>(Cxy, Cyx, fpot, gpot,
      MW, ML,  1.0f, scalarOut, cnt, gen);
  sinkhorn_kernel<<<dim3(SBLOCKS), dim3(256), 0, stream>>>(Cxx, Cxx, fpot, gpot,
      MW, MW, -0.5f, scalarOut, cnt, gen);
  sinkhorn_kernel<<<dim3(SBLOCKS), dim3(256), 0, stream>>>(Cyy, Cyy, fpot, gpot,
      ML, ML, -0.5f, scalarOut, cnt, gen);

  // 5) aligned logits = E1 @ W_out + b_out  (full-K per block, bias fused)
  gemm_bf16_kernel<<<dim3((VOUT + BLK_N - 1)/BLK_N, MW/BLK_M, 1),
                    dim3(GTHREADS), GEMM_LDS_BYTES, stream>>>(
      E1, Wo, bo, out, MW, VOUT, DPROJ, DPROJ, 0);
}
